// Intention_1219770712463
// MI455X (gfx1250) — matrix operand order repaired and checked by reading, not compiled, not run on hardware
//
#include <hip/hip_runtime.h>
#include <hip/hip_bf16.h>

typedef __bf16 bf16;
typedef __attribute__((ext_vector_type(16))) __bf16 v16bf;
typedef __attribute__((ext_vector_type(8)))  __bf16 v8bf;
typedef __attribute__((ext_vector_type(8)))  float  v8f;

#define WMMA_BF16(a, b, c) \
  __builtin_amdgcn_wmma_f32_16x16x32_bf16(false, (a), false, (b), (short)0, (c), false, false)

// CDNA5 async global->LDS copy support (ASYNCcnt path).  Guarded so the build
// falls back to VGPR staging if this toolchain lacks the builtin.
#if defined(__HIP_DEVICE_COMPILE__) && \
    __has_builtin(__builtin_amdgcn_global_load_async_to_lds_b128)
#define HAVE_ASYNC_LDS 1
#endif

#if defined(HAVE_ASYNC_LDS)
// Exact parameter type per hipcc diagnostic: int __attribute__((vector_size(16)))*
typedef int v4i_vs __attribute__((vector_size(16)));
typedef __attribute__((address_space(1))) v4i_vs* g_v4i_ptr;   // global src
typedef __attribute__((address_space(3))) v4i_vs* l_v4i_ptr;   // LDS dst
#endif

// Problem constants
constexpr int Bsz = 2;
constexpr int S   = 2048;
constexpr int D   = 1024;
constexpr int NH  = 16;
constexpr int HD  = 64;
constexpr int MS  = Bsz * S;   // 4096 rows total
constexpr int BH  = Bsz * NH;  // 32 (batch*head)

// ---------------------------------------------------------------------------
// 16B global -> LDS copy.  Async DMA (no VGPR round-trip, ASYNCcnt) when
// available; plain uint4 staging otherwise.
// ---------------------------------------------------------------------------
__device__ __forceinline__ void g2lds_b128(const bf16* gsrc, bf16* ldst) {
#if defined(HAVE_ASYNC_LDS)
  // AS1 ptr == flat VA bits; AS3 ptr == low 32 bits of generic LDS address
  // (ISA 10.2).
  __builtin_amdgcn_global_load_async_to_lds_b128(
      (g_v4i_ptr)(uintptr_t)gsrc,
      (l_v4i_ptr)(uint32_t)(uintptr_t)ldst,
      0, 0);
#else
  *(uint4*)ldst = *(const uint4*)gsrc;
#endif
}

// Drain this wave's async copies, then workgroup barrier so every wave sees
// every tile element.
__device__ __forceinline__ void wait_async_and_barrier() {
#if defined(HAVE_ASYNC_LDS)
#if __has_builtin(__builtin_amdgcn_s_wait_asynccnt)
  __builtin_amdgcn_s_wait_asynccnt(0);
#else
  asm volatile("s_wait_asynccnt 0x0" ::: "memory");
#endif
#endif
  __syncthreads();
}

// ---------------------------------------------------------------------------
// A-fragment loader: 16x32 bf16 A matrix, row-major source, rowbase points at
// element (m, k0). Lane layout: M = lane%16, element j -> K = half*16 + j.
// Two 16B loads.
// ---------------------------------------------------------------------------
__device__ __forceinline__ v16bf ldA16(const bf16* rowbase, int half) {
  v8bf lo = *(const v8bf*)(rowbase + half * 16);
  v8bf hi = *(const v8bf*)(rowbase + half * 16 + 8);
  v16bf r;
#pragma unroll
  for (int i = 0; i < 8; ++i) { r[i] = lo[i]; r[i + 8] = hi[i]; }
  return r;
}

// ---------------------------------------------------------------------------
// fp32 -> bf16 conversion
// ---------------------------------------------------------------------------
__global__ void cvt_f32_bf16(const float* __restrict__ src,
                             bf16* __restrict__ dst, int n) {
  int i = blockIdx.x * blockDim.x + threadIdx.x;
  if (i < n) dst[i] = (bf16)src[i];
}

// ---------------------------------------------------------------------------
// Generic NT GEMM:  C[M,N] = A[M,K] * B[N,K]^T + bias[N]
// A,B bf16 row-major (K contiguous).  Optional fp32 output Cf (row-major) and
// bf16 output Cb.  split!=0 scatters Cb into head-split [B,NH,S,HD] layout.
// Block tile 128(M) x 128(N) x 32(K); 8 waves as 4(M) x 2(N), each wave
// computes 32x64 = 2x4 WMMA tiles -> 8 wmma per 12 ds_load_b128 per k-step.
// Tile fill uses async global->LDS DMA (ASYNCcnt) when available.
// ---------------------------------------------------------------------------
__global__ __launch_bounds__(256) void gemm_bf16_nt(
    const bf16* __restrict__ A, const bf16* __restrict__ Bm,
    const float* __restrict__ bias, float* __restrict__ Cf,
    bf16* __restrict__ Cb, int M, int N, int K, int split) {
  __shared__ bf16 As[128 * 32];
  __shared__ bf16 Bs[128 * 32];

  const int tid  = threadIdx.x;
  const int nblk = blockIdx.x * 128;
  const int mblk = blockIdx.y * 128;
  const int wave = tid >> 5;
  const int lane = tid & 31;
  const int half = lane >> 4;
  const int l15  = lane & 15;
  const int wm   = wave & 3;   // 4 waves along M (32 rows each)
  const int wn   = wave >> 2;  // 2 waves along N (64 cols each)
  const int m0   = wm * 32;
  const int n0   = wn * 64;

  // Per-thread staging coordinates (2 chunks of 8 halfs for A and for B).
  const int c0 = tid * 2, c1 = tid * 2 + 1;
  const int row0 = c0 >> 2, off0 = (c0 & 3) * 8;
  const int row1 = c1 >> 2, off1 = (c1 & 3) * 8;

  v8f acc[2][4] = {{{}, {}, {}, {}}, {{}, {}, {}, {}}};

  for (int kb = 0; kb < K; kb += 32) {
    __syncthreads();  // previous k-step's readers done before overwrite
    g2lds_b128(A + (size_t)(mblk + row0) * K + kb + off0, As + row0 * 32 + off0);
    g2lds_b128(A + (size_t)(mblk + row1) * K + kb + off1, As + row1 * 32 + off1);
    g2lds_b128(Bm + (size_t)(nblk + row0) * K + kb + off0, Bs + row0 * 32 + off0);
    g2lds_b128(Bm + (size_t)(nblk + row1) * K + kb + off1, Bs + row1 * 32 + off1);
    if (kb + 32 < K) {  // hint next k-slab into cache (global_prefetch_b8)
      __builtin_prefetch(A + (size_t)(mblk + row0) * K + kb + 32, 0, 1);
      __builtin_prefetch(Bm + (size_t)(nblk + row0) * K + kb + 32, 0, 1);
    }
    wait_async_and_barrier();

    v16bf af[2], bfr[4];
#pragma unroll
    for (int i = 0; i < 2; ++i)
      af[i] = ldA16(As + (m0 + i * 16 + l15) * 32, half);
#pragma unroll
    for (int j = 0; j < 4; ++j)
      bfr[j] = *(const v16bf*)(Bs + (n0 + j * 16 + l15) * 32 + half * 16);
#pragma unroll
    for (int i = 0; i < 2; ++i)
#pragma unroll
      for (int j = 0; j < 4; ++j)
        acc[i][j] = WMMA_BF16(af[i], bfr[j], acc[i][j]);
  }

  // Epilogue: C layout lane(M = r + 8*half, N = l15)
#pragma unroll
  for (int i = 0; i < 2; ++i) {
#pragma unroll
    for (int j = 0; j < 4; ++j) {
      int gcol = nblk + n0 + j * 16 + l15;
      float bv = bias ? bias[gcol] : 0.f;
#pragma unroll
      for (int r = 0; r < 8; ++r) {
        int grow = mblk + m0 + i * 16 + r + 8 * half;
        float val = acc[i][j][r] + bv;
        if (Cf) Cf[(size_t)grow * N + gcol] = val;
        if (Cb) {
          size_t dst;
          if (split) {  // [B,S,D] -> [B,NH,S,HD]
            int b = grow >> 11, s = grow & 2047;
            int hidx = gcol >> 6, hd = gcol & 63;
            dst = (((size_t)(b * NH + hidx) * S) + s) * HD + hd;
          } else {
            dst = (size_t)grow * N + gcol;
          }
          Cb[dst] = (bf16)val;
        }
      }
    }
  }
}

// ---------------------------------------------------------------------------
// Gram: gram[h] = K_h^T K_h  (64x64 per head), K_h = [2048,64] bf16.
// One workgroup (8 waves) per head; 16 output tiles, 2 per wave.
// A[m][k] = K[k][m], B[n][k] = K[k][n] (strided gathers; kernel is tiny).
// ---------------------------------------------------------------------------
__global__ __launch_bounds__(256) void gram64(const bf16* __restrict__ Kb,
                                              float* __restrict__ gram) {
  const int h = blockIdx.x;
  const bf16* Kh = Kb + (size_t)h * S * HD;
  const int tid = threadIdx.x;
  const int wave = tid >> 5, lane = tid & 31, half = lane >> 4, l15 = lane & 15;

  for (int t = wave * 2; t < wave * 2 + 2; ++t) {
    int m0 = (t >> 2) * 16, n0 = (t & 3) * 16;
    v8f acc = {};
    for (int s0 = 0; s0 < S; s0 += 32) {
      v16bf a, b;
#pragma unroll
      for (int j = 0; j < 16; ++j) {
        int ka = half * 16 + j;
        a[j] = Kh[(size_t)(s0 + ka) * HD + m0 + l15];
        int kk = half * 16 + j;
        b[j] = Kh[(size_t)(s0 + kk) * HD + n0 + l15];
      }
      acc = WMMA_BF16(a, b, acc);
    }
#pragma unroll
    for (int r = 0; r < 8; ++r)
      gram[(size_t)h * 4096 + (m0 + r + 8 * half) * 64 + n0 + l15] = acc[r];
  }
}

// ---------------------------------------------------------------------------
// 64x64 Gauss-Jordan inverse per head (fp32 in LDS; gram is SPD ~2048*I so no
// pivoting needed).  Output bf16 (symmetric, so it doubles as B[n][k]).
// ---------------------------------------------------------------------------
__global__ __launch_bounds__(128) void invert64(const float* __restrict__ gram,
                                                bf16* __restrict__ invb) {
  __shared__ float Mx[64][130];
  __shared__ float fac[64];
  const int h = blockIdx.x;
  const float* g = gram + (size_t)h * 4096;
  const int c = threadIdx.x;  // 0..127 : one augmented column per thread

  for (int r = 0; r < 64; ++r)
    Mx[r][c] = (c < 64) ? g[r * 64 + c] : ((c - 64) == r ? 1.f : 0.f);
  __syncthreads();

  for (int k = 0; k < 64; ++k) {
    float rpv = 1.f / Mx[k][k];
    __syncthreads();
    Mx[k][c] *= rpv;
    __syncthreads();
    if (c < 64) fac[c] = Mx[c][k];
    __syncthreads();
    float pkc = Mx[k][c];
    for (int r = 0; r < 64; ++r)
      if (r != k) Mx[r][c] -= fac[r] * pkc;
    __syncthreads();
  }
  if (c >= 64)
    for (int r = 0; r < 64; ++r)
      invb[(size_t)h * 4096 + r * 64 + (c - 64)] = (bf16)Mx[r][c];
}

// ---------------------------------------------------------------------------
// q_proj = Q_h * Ginv_h  per head.  M=2048, N=64, K=64; Ginv symmetric so it
// is directly the NT B operand.  8 waves x 16 rows, 4 n-tiles each.
// ---------------------------------------------------------------------------
__global__ __launch_bounds__(256) void qproj64(const bf16* __restrict__ Qb,
                                               const bf16* __restrict__ invb,
                                               bf16* __restrict__ QPb) {
  const int h = blockIdx.y;
  const int mblk = blockIdx.x * 128;
  const bf16* Qh = Qb + (size_t)h * S * HD;
  const bf16* Ih = invb + (size_t)h * 4096;
  bf16* Oh = QPb + (size_t)h * S * HD;
  const int tid = threadIdx.x;
  const int wave = tid >> 5, lane = tid & 31, half = lane >> 4, l15 = lane & 15;
  const int m0 = mblk + wave * 16;

  v8f acc[4] = {{}, {}, {}, {}};
#pragma unroll
  for (int k0 = 0; k0 < 64; k0 += 32) {
    v16bf a = ldA16(Qh + (size_t)(m0 + l15) * HD + k0, half);
#pragma unroll
    for (int j = 0; j < 4; ++j) {
      v16bf b = *(const v16bf*)(Ih + (size_t)(j * 16 + l15) * 64 + k0 + half * 16);
      acc[j] = WMMA_BF16(a, b, acc[j]);
    }
  }
#pragma unroll
  for (int j = 0; j < 4; ++j)
#pragma unroll
    for (int r = 0; r < 8; ++r)
      Oh[(size_t)(m0 + r + 8 * half) * HD + j * 16 + l15] = (bf16)acc[j][r];
}

// ---------------------------------------------------------------------------
// Fused attention (flash style): per (head, 128-row query tile), 8 waves loop
// over key tiles of 32.  score = q_proj * K^T (WMMA, K=64), online softmax
// (scale 1/32 = 1/sqrt(1024)), P*V (WMMA, K=32) with P transposed to A-layout
// via per-wave LDS and V staged in shared LDS via async DMA (amortized over
// 8 waves).  Never materializes the 2048x2048 score matrix.
// ---------------------------------------------------------------------------
__global__ __launch_bounds__(256) void flash_attn(
    const bf16* __restrict__ QPb, const bf16* __restrict__ Kb,
    const bf16* __restrict__ Vb, bf16* __restrict__ AOb) {
  const int h = blockIdx.y;
  const int b = h >> 4, nh = h & 15;
  const bf16* Qh = QPb + (size_t)h * S * HD;
  const bf16* Kh = Kb + (size_t)h * S * HD;
  const bf16* Vh = Vb + (size_t)h * S * HD;
  const int tid = threadIdx.x;
  const int wave = tid >> 5, lane = tid & 31, half = lane >> 4, l15 = lane & 15;
  const int m0 = blockIdx.x * 128 + wave * 16;

  __shared__ bf16 Vlds[32 * 64];       // key-tile of V, shared by 8 waves
  __shared__ bf16 Plds[8][16 * 32];    // per-wave P transpose staging

  // q_proj fragments for this wave's 16 rows (K=64 -> two frags), loaded once.
  const bf16* pa = Qh + (size_t)(m0 + l15) * HD;
  v16bf aq0 = ldA16(pa, half);
  v16bf aq1 = ldA16(pa + 32, half);

  v8f acc[4] = {{}, {}, {}, {}};
  float mrow[8], lrow[8];
#pragma unroll
  for (int r = 0; r < 8; ++r) { mrow[r] = -3.0e38f; lrow[r] = 0.f; }

  for (int t = 0; t < S / 32; ++t) {
    const int kv0 = t * 32;
    __syncthreads();  // previous iteration's Vlds readers done
    // V tile: 32x64 bf16 = 256 16B chunks, 1 per thread, async to LDS.
    {
      int row = tid >> 3, off = (tid & 7) * 8;
      g2lds_b128(Vh + (size_t)(kv0 + row) * HD + off, Vlds + row * 64 + off);
    }
    wait_async_and_barrier();

    // Scores: two 16x16 key sub-tiles, K=64 each.
    v8f sc0 = {}, sc1 = {};
    {
      const bf16* pb0 = Kh + (size_t)(kv0 + l15) * HD + half * 16;
      const bf16* pb1 = Kh + (size_t)(kv0 + 16 + l15) * HD + half * 16;
      v16bf b00 = *(const v16bf*)(pb0);
      v16bf b01 = *(const v16bf*)(pb0 + 32);
      v16bf b10 = *(const v16bf*)(pb1);
      v16bf b11 = *(const v16bf*)(pb1 + 32);
      sc0 = WMMA_BF16(aq0, b00, sc0);
      sc0 = WMMA_BF16(aq1, b01, sc0);
      sc1 = WMMA_BF16(aq0, b10, sc1);
      sc1 = WMMA_BF16(aq1, b11, sc1);
    }

    // Online softmax (row = one VGPR across the 16 lanes of each half).
#pragma unroll
    for (int r = 0; r < 8; ++r) {
      float s0 = sc0[r] * 0.03125f;  // 1/sqrt(EMBED_DIM) = 1/32
      float s1 = sc1[r] * 0.03125f;
      float mx = fmaxf(s0, s1);
      mx = fmaxf(mx, __shfl_xor(mx, 1));
      mx = fmaxf(mx, __shfl_xor(mx, 2));
      mx = fmaxf(mx, __shfl_xor(mx, 4));
      mx = fmaxf(mx, __shfl_xor(mx, 8));
      float mnew = fmaxf(mrow[r], mx);
      float scale = __expf(mrow[r] - mnew);
      float p0 = __expf(s0 - mnew);
      float p1 = __expf(s1 - mnew);
      float sum = p0 + p1;
      sum += __shfl_xor(sum, 1);
      sum += __shfl_xor(sum, 2);
      sum += __shfl_xor(sum, 4);
      sum += __shfl_xor(sum, 8);
      lrow[r] = lrow[r] * scale + sum;
      mrow[r] = mnew;
#pragma unroll
      for (int jn = 0; jn < 4; ++jn) acc[jn][r] *= scale;
      Plds[wave][(r + 8 * half) * 32 + l15] = (bf16)p0;
      Plds[wave][(r + 8 * half) * 32 + 16 + l15] = (bf16)p1;
    }

    // P (16x32) * V (32x64): P back as A-frag; V gathered transposed from LDS.
    v16bf ap = ldA16(&Plds[wave][l15 * 32], half);
#pragma unroll
    for (int jn = 0; jn < 4; ++jn) {
      v16bf bv;
#pragma unroll
      for (int j = 0; j < 16; ++j)
        bv[j] = Vlds[(half * 16 + j) * 64 + jn * 16 + l15];
      acc[jn] = WMMA_BF16(ap, bv, acc[jn]);
    }
  }

  // Normalize and scatter to merged-head [B*S, D] bf16.
#pragma unroll
  for (int r = 0; r < 8; ++r) {
    float rl = 1.f / lrow[r];
    int s = m0 + r + 8 * half;
    size_t rowbase = ((size_t)b * S + s) * D + nh * HD;
#pragma unroll
    for (int jn = 0; jn < 4; ++jn)
      AOb[rowbase + jn * 16 + l15] = (bf16)(acc[jn][r] * rl);
  }
}

// ---------------------------------------------------------------------------
// Host-side orchestration
// ---------------------------------------------------------------------------
extern "C" void kernel_launch(void* const* d_in, const int* in_sizes, int n_in,
                              void* d_out, int out_size, void* d_ws, size_t ws_size,
                              hipStream_t stream) {
  (void)in_sizes; (void)n_in; (void)out_size; (void)ws_size;
  const float* x   = (const float*)d_in[0];
  const float* Wq  = (const float*)d_in[1];
  const float* bq  = (const float*)d_in[2];
  const float* Wk  = (const float*)d_in[3];
  const float* bk  = (const float*)d_in[4];
  const float* Wv  = (const float*)d_in[5];
  const float* bv  = (const float*)d_in[6];
  const float* Wfc = (const float*)d_in[7];
  const float* bfc = (const float*)d_in[8];

  char* ws = (char*)d_ws;
  size_t off = 0;
  auto alloc = [&](size_t bytes) -> void* {
    void* p = ws + off;
    off += (bytes + 255) & ~(size_t)255;
    return p;
  };
  bf16*  xb   = (bf16*)alloc((size_t)MS * D * 2);
  bf16*  Wqb  = (bf16*)alloc((size_t)D * D * 2);
  bf16*  Wkb  = (bf16*)alloc((size_t)D * D * 2);
  bf16*  Wvb  = (bf16*)alloc((size_t)D * D * 2);
  bf16*  Wfcb = (bf16*)alloc((size_t)D * D * 2);
  bf16*  Qb   = (bf16*)alloc((size_t)BH * S * HD * 2);
  bf16*  Kbp  = (bf16*)alloc((size_t)BH * S * HD * 2);
  bf16*  Vbp  = (bf16*)alloc((size_t)BH * S * HD * 2);
  float* gram = (float*)alloc((size_t)BH * 64 * 64 * 4);
  bf16*  invb = (bf16*)alloc((size_t)BH * 64 * 64 * 2);
  bf16*  QPb  = (bf16*)alloc((size_t)BH * S * HD * 2);
  bf16*  AOb  = (bf16*)alloc((size_t)MS * D * 2);

  // 1) fp32 -> bf16 conversions
  const int nX = MS * D, nW = D * D;
  cvt_f32_bf16<<<(nX + 255) / 256, 256, 0, stream>>>(x, xb, nX);
  cvt_f32_bf16<<<(nW + 255) / 256, 256, 0, stream>>>(Wq, Wqb, nW);
  cvt_f32_bf16<<<(nW + 255) / 256, 256, 0, stream>>>(Wk, Wkb, nW);
  cvt_f32_bf16<<<(nW + 255) / 256, 256, 0, stream>>>(Wv, Wvb, nW);
  cvt_f32_bf16<<<(nW + 255) / 256, 256, 0, stream>>>(Wfc, Wfcb, nW);

  // 2) Q/K/V projections (head-split bf16 outputs)
  dim3 gg(D / 128, MS / 128);
  gemm_bf16_nt<<<gg, 256, 0, stream>>>(xb, Wqb, bq, nullptr, Qb, MS, D, D, 1);
  gemm_bf16_nt<<<gg, 256, 0, stream>>>(xb, Wkb, bk, nullptr, Kbp, MS, D, D, 1);
  gemm_bf16_nt<<<gg, 256, 0, stream>>>(xb, Wvb, bv, nullptr, Vbp, MS, D, D, 1);

  // 3) Gram matrices + inverses
  gram64<<<BH, 256, 0, stream>>>(Kbp, gram);
  invert64<<<BH, 128, 0, stream>>>(gram, invb);

  // 4) q_proj = Q * Ginv
  qproj64<<<dim3(S / 128, BH), 256, 0, stream>>>(Qb, invb, QPb);

  // 5) fused score/softmax/PV
  flash_attn<<<dim3(S / 128, BH), 256, 0, stream>>>(QPb, Kbp, Vbp, AOb);

  // 6) final FC -> fp32 d_out
  gemm_bf16_nt<<<gg, 256, 0, stream>>>(AOb, Wfcb, bfc, (float*)d_out, nullptr,
                                       MS, D, D, 0);
}
